// GINLayer_55765855371637
// MI455X (gfx1250) — matrix operand order repaired and checked by reading, not compiled, not run on hardware
//
#include <hip/hip_runtime.h>
#include <hip/hip_bf16.h>

#define NN 100000
#define EE 1600000
#define DD 128
#define HH 256
#define BN_EPS 1e-5f

typedef __attribute__((ext_vector_type(16))) __bf16 v16bf;
typedef __attribute__((ext_vector_type(8)))  float  v8f;

// ---------------- small helpers ----------------

__global__ void zero_stats_kernel(float* __restrict__ st, int n) {
    int i = blockIdx.x * blockDim.x + threadIdx.x;
    if (i < n) st[i] = 0.0f;
}

__global__ void init_h_kernel(const float* __restrict__ x,
                              const float* __restrict__ eps,
                              float* __restrict__ h) {
    int i = blockIdx.x * blockDim.x + threadIdx.x;
    if (i < NN * DD) h[i] = (1.0f + eps[0]) * x[i];
}

// one thread per (edge, 4-float chunk): 32 threads per edge cover D=128
__global__ void scatter_kernel(const float* __restrict__ x,
                               const int* __restrict__ src,
                               const int* __restrict__ dst,
                               float* __restrict__ h) {
    long long gid = (long long)blockIdx.x * blockDim.x + threadIdx.x;
    if (gid >= (long long)EE * 32) return;
    int e = (int)(gid >> 5);
    int f = ((int)gid & 31) << 2;
    int s = src[e];
    int d = dst[e];
    float4 v = *reinterpret_cast<const float4*>(x + (long long)s * DD + f);
    float* p = h + (long long)d * DD + f;
    atomicAdd(p + 0, v.x);
    atomicAdd(p + 1, v.y);
    atomicAdd(p + 2, v.z);
    atomicAdd(p + 3, v.w);
}

// per-column sum & sum-of-squares over a row chunk, atomically merged
__global__ void colstats_kernel(const float* __restrict__ u, int ncols,
                                int rows_per_block,
                                float* __restrict__ sum,
                                float* __restrict__ sumsq) {
    int col = threadIdx.x;
    int r0 = blockIdx.x * rows_per_block;
    int r1 = r0 + rows_per_block;
    if (r1 > NN) r1 = NN;
    float s = 0.0f, q = 0.0f;
    for (int r = r0; r < r1; ++r) {
        float v = u[(long long)r * ncols + col];
        s += v;
        q += v * v;
    }
    atomicAdd(&sum[col], s);
    atomicAdd(&sumsq[col], q);
}

// BN affine coefficients: a = gamma*rsqrt(var+eps), c = beta - mu*a
__global__ void coef_kernel(const float* __restrict__ sum,
                            const float* __restrict__ sumsq,
                            const float* __restrict__ gamma,
                            const float* __restrict__ beta,
                            int ncols, float* __restrict__ a,
                            float* __restrict__ c) {
    int t = blockIdx.x * blockDim.x + threadIdx.x;
    if (t < ncols) {
        float inv_n = 1.0f / (float)NN;
        float mu = sum[t] * inv_n;
        float var = sumsq[t] * inv_n - mu * mu;
        float s = gamma[t] * rsqrtf(var + BN_EPS);
        a[t] = s;
        c[t] = beta[t] - mu * s;
    }
}

// out = relu(a2[col]*u2 + c2[col] + x)   (u2 lives in `out` as scratch)
__global__ void final_kernel(const float* __restrict__ x,
                             const float* __restrict__ a2,
                             const float* __restrict__ c2,
                             float* __restrict__ out) {
    int i = blockIdx.x * blockDim.x + threadIdx.x;
    if (i < NN * DD) {
        int col = i & (DD - 1);
        float v = out[i] * a2[col] + c2[col] + x[i];
        out[i] = fmaxf(v, 0.0f);
    }
}

// ---------------- WMMA GEMMs ----------------
// One wave computes a 16x16 f32 tile; K stepped 32 at a time via
// v_wmma_f32_16x16x32_bf16. Layouts per CDNA5 ISA §7.12.2.

__device__ __forceinline__ v16bf pack_a16(const float av[16]) {
    v16bf a;
#pragma unroll
    for (int j = 0; j < 16; ++j) a[j] = (__bf16)av[j];
    return a;
}

// u1[N,H] = h[N,D] @ W1[D,H] + b1
__global__ __launch_bounds__(256) void gemm1_kernel(
    const float* __restrict__ h, const float* __restrict__ W1,
    const float* __restrict__ b1, float* __restrict__ u1) {
    int wave = threadIdx.x >> 5;
    int lane = threadIdx.x & 31;
    int m0 = blockIdx.x * 16;
    int n0 = (blockIdx.y * 8 + wave) * 16;
    int mrow = m0 + (lane & 15);
    int half = lane >> 4;

    v8f acc = {};
#pragma unroll
    for (int kk = 0; kk < 4; ++kk) {
        // A: row mrow, K = kk*32 + {half*8+0..7, 16+half*8+0..7}
        const float* ap = h + (long long)mrow * DD + kk * 32 + half * 8;
        float4 p0 = *(const float4*)(ap + 0);
        float4 p1 = *(const float4*)(ap + 4);
        float4 q0 = *(const float4*)(ap + 16);
        float4 q1 = *(const float4*)(ap + 20);
        float av[16] = {p0.x, p0.y, p0.z, p0.w, p1.x, p1.y, p1.z, p1.w,
                        q0.x, q0.y, q0.z, q0.w, q1.x, q1.y, q1.z, q1.w};
        v16bf a = pack_a16(av);

        // B: col n0+(lane&15), K = kk*32 + {half*8+0..7, 16+half*8+0..7}
        const float* bp = W1 + (long long)(kk * 32 + half * 8) * HH + n0 + (lane & 15);
        float bv[16];
#pragma unroll
        for (int j = 0; j < 8; ++j) {
            bv[j]     = bp[(long long)j * HH];
            bv[8 + j] = bp[(long long)(16 + j) * HH];
        }
        v16bf b = pack_a16(bv);

        acc = __builtin_amdgcn_wmma_f32_16x16x32_bf16(
            false, a, false, b, (short)0, acc, false, false);
    }

    int n = n0 + (lane & 15);
    float bias = b1[n];
#pragma unroll
    for (int r = 0; r < 8; ++r) {
        int m = m0 + r + half * 8;
        u1[(long long)m * HH + n] = acc[r] + bias;
    }
}

// out[N,D] = relu(a1*u1 + c1)[N,H] @ W2[H,D] + b2   (BN1+ReLU fused into A)
__global__ __launch_bounds__(256) void gemm2_kernel(
    const float* __restrict__ u1, const float* __restrict__ a1,
    const float* __restrict__ c1, const float* __restrict__ W2,
    const float* __restrict__ b2, float* __restrict__ out) {
    __shared__ float sa[HH];
    __shared__ float sc[HH];
    sa[threadIdx.x] = a1[threadIdx.x];
    sc[threadIdx.x] = c1[threadIdx.x];
    __syncthreads();

    int wave = threadIdx.x >> 5;   // 8 waves cover all 8 N-tiles (D=128)
    int lane = threadIdx.x & 31;
    int m0 = blockIdx.x * 16;
    int n0 = wave * 16;
    int mrow = m0 + (lane & 15);
    int half = lane >> 4;

    v8f acc = {};
#pragma unroll
    for (int kk = 0; kk < 8; ++kk) {
        int cb = kk * 32 + half * 8;
        const float* ap = u1 + (long long)mrow * HH + cb;
        float4 p0 = *(const float4*)(ap + 0);
        float4 p1 = *(const float4*)(ap + 4);
        float4 q0 = *(const float4*)(ap + 16);
        float4 q1 = *(const float4*)(ap + 20);
        float raw[16] = {p0.x, p0.y, p0.z, p0.w, p1.x, p1.y, p1.z, p1.w,
                         q0.x, q0.y, q0.z, q0.w, q1.x, q1.y, q1.z, q1.w};
        float av[16];
#pragma unroll
        for (int j = 0; j < 8; ++j) {
            av[j]     = fmaxf(raw[j]     * sa[cb + j]      + sc[cb + j],      0.0f);
            av[8 + j] = fmaxf(raw[8 + j] * sa[cb + 16 + j] + sc[cb + 16 + j], 0.0f);
        }
        v16bf a = pack_a16(av);

        const float* bp = W2 + (long long)cb * DD + n0 + (lane & 15);
        float bv[16];
#pragma unroll
        for (int j = 0; j < 8; ++j) {
            bv[j]     = bp[(long long)j * DD];
            bv[8 + j] = bp[(long long)(16 + j) * DD];
        }
        v16bf b = pack_a16(bv);

        acc = __builtin_amdgcn_wmma_f32_16x16x32_bf16(
            false, a, false, b, (short)0, acc, false, false);
    }

    int n = n0 + (lane & 15);
    float bias = b2[n];
#pragma unroll
    for (int r = 0; r < 8; ++r) {
        int m = m0 + r + half * 8;
        out[(long long)m * DD + n] = acc[r] + bias;
    }
}

// ---------------- launch ----------------

extern "C" void kernel_launch(void* const* d_in, const int* in_sizes, int n_in,
                              void* d_out, int out_size, void* d_ws, size_t ws_size,
                              hipStream_t stream) {
    (void)in_sizes; (void)n_in; (void)out_size; (void)ws_size;

    const float* x   = (const float*)d_in[0];
    const int*   ei  = (const int*)d_in[1];   // [2, E] int32
    const float* eps = (const float*)d_in[2];
    const float* W1  = (const float*)d_in[3];
    const float* b1  = (const float*)d_in[4];
    const float* g1  = (const float*)d_in[5];
    const float* be1 = (const float*)d_in[6];
    const float* W2  = (const float*)d_in[7];
    const float* b2  = (const float*)d_in[8];
    const float* g2  = (const float*)d_in[9];
    const float* be2 = (const float*)d_in[10];
    float* out = (float*)d_out;

    char* ws = (char*)d_ws;
    float* h  = (float*)ws;                                           // N*D
    float* u1 = (float*)(ws + (size_t)NN * DD * sizeof(float));       // N*H
    float* st = (float*)(ws + (size_t)NN * DD * sizeof(float)
                            + (size_t)NN * HH * sizeof(float));
    float* sum1 = st;                  // HH
    float* sq1  = st + HH;             // HH
    float* a1   = st + 2 * HH;         // HH
    float* c1   = st + 3 * HH;         // HH
    float* sum2 = st + 4 * HH;         // DD
    float* sq2  = st + 4 * HH + DD;    // DD
    float* a2   = st + 4 * HH + 2*DD;  // DD
    float* c2   = st + 4 * HH + 3*DD;  // DD

    const int* srcv = ei;       // edge_index[0]
    const int* dstv = ei + EE;  // edge_index[1]

    int nstats = 4 * HH + 4 * DD;
    zero_stats_kernel<<<(nstats + 255) / 256, 256, 0, stream>>>(st, nstats);

    init_h_kernel<<<(NN * DD + 255) / 256, 256, 0, stream>>>(x, eps, h);

    long long sthreads = (long long)EE * 32;
    scatter_kernel<<<(int)((sthreads + 255) / 256), 256, 0, stream>>>(x, srcv, dstv, h);

    dim3 grid1(NN / 16, HH / (16 * 8));  // (6250, 2), 8 waves/block
    gemm1_kernel<<<grid1, 256, 0, stream>>>(h, W1, b1, u1);

    colstats_kernel<<<(NN + 249) / 250, HH, 0, stream>>>(u1, HH, 250, sum1, sq1);
    coef_kernel<<<1, HH, 0, stream>>>(sum1, sq1, g1, be1, HH, a1, c1);

    gemm2_kernel<<<NN / 16, 256, 0, stream>>>(u1, a1, c1, W2, b2, out);

    colstats_kernel<<<(NN + 249) / 250, DD, 0, stream>>>(out, DD, 250, sum2, sq2);
    coef_kernel<<<1, DD, 0, stream>>>(sum2, sq2, g2, be2, DD, a2, c2);

    final_kernel<<<(NN * DD + 255) / 256, 256, 0, stream>>>(x, a2, c2, out);
}
